// AttentionBlock_7318624272690
// MI455X (gfx1250) — matrix operand order repaired and checked by reading, not compiled, not run on hardware
//
#include <hip/hip_runtime.h>
#include <hip/hip_bf16.h>
#include <math.h>

// ---------------------------------------------------------------------------
// AttentionBlock for MI455X (gfx1250): GroupNorm + QKV + flash attention +
// out-proj, matmuls via v_wmma_f32_16x16x32_f16 (wave32).
// ---------------------------------------------------------------------------

typedef _Float16 half_t;
typedef __attribute__((ext_vector_type(16))) _Float16 v16h;
typedef __attribute__((ext_vector_type(8)))  _Float16 v8h;
typedef __attribute__((ext_vector_type(8)))  float    v8f;

constexpr int kB   = 16;
constexpr int kC   = 512;
constexpr int kN   = 1024;           // H*W = 32*32
constexpr int kG   = 32;             // groups
constexpr int kCPG = kC / kG;        // 16 channels per group
constexpr int kO3  = 3 * kC;         // 1536
constexpr float kScale = 0.044194173824159216f;  // 1/sqrt(512)
constexpr float kEps = 1e-5f;

// ---------------- WMMA helpers (contiguous k order, same in A and B) ---------------

__device__ __forceinline__ v8f wmma_f16(v16h a, v16h b, v8f c) {
  return __builtin_amdgcn_wmma_f32_16x16x32_f16(
      /*neg_a=*/false, a, /*neg_b=*/false, b,
      /*c_mod=*/(short)0, c, /*reuse_a=*/false, /*reuse_b=*/false);
}

// A fragment: 16x32 f16 tile, `row` points at this lane's matrix row (row-
// major, K contiguous). lanes<16: elems 0..15=k0..15;
// lanes>=16: k16..31.
__device__ __forceinline__ v16h load_frag_a(const half_t* row, int kbase, int hi) {
  return *reinterpret_cast<const v16h*>(row + kbase + hi * 16);
}

// B fragment: 32x16 f16 tile stored [col][k] (k contiguous). `col_row` points
// at this lane's column (= lane%16); lanes<16 take k0..15, lanes>=16 k16..31.
__device__ __forceinline__ v16h load_frag_b(const half_t* col_row, int kbase, int hi) {
  return *reinterpret_cast<const v16h*>(col_row + kbase + hi * 16);
}

// ----------------------------- kernel 1: GN stats ---------------------------

__global__ __launch_bounds__(256)
void gn_stats_kernel(const float* __restrict__ x,
                     float* __restrict__ mean, float* __restrict__ rstd) {
  __shared__ float ssum[256];
  __shared__ float ssq[256];
  const int tid = threadIdx.x;
  const int bid = blockIdx.x;               // b*G + g
  const int b = bid >> 5, g = bid & 31;
  const size_t base = (size_t)b * kC * kN + (size_t)g * kCPG * kN;
  const int count = kCPG * kN;              // 16384
  float s = 0.f, q = 0.f;
  for (int i = tid; i < count; i += 256) {
    float v = x[base + i];
    s += v; q += v * v;
  }
  ssum[tid] = s; ssq[tid] = q;
  __syncthreads();
  for (int off = 128; off > 0; off >>= 1) {
    if (tid < off) { ssum[tid] += ssum[tid + off]; ssq[tid] += ssq[tid + off]; }
    __syncthreads();
  }
  if (tid == 0) {
    float m = ssum[0] / count;
    float var = ssq[0] / count - m * m;
    mean[bid] = m;
    rstd[bid] = rsqrtf(var + kEps);
  }
}

// ------------------- kernel 2: GN apply + transpose to [b][n][c] ------------

__global__ __launch_bounds__(256)
void gn_apply_kernel(const float* __restrict__ x,
                     const float* __restrict__ gn_w, const float* __restrict__ gn_b,
                     const float* __restrict__ mean, const float* __restrict__ rstd,
                     half_t* __restrict__ h_bnc) {
  const size_t idx = (size_t)blockIdx.x * 256 + threadIdx.x;  // (b,n,c) c-fastest
  const int c = (int)(idx & (kC - 1));
  const int n = (int)((idx >> 9) & (kN - 1));
  const int b = (int)(idx >> 19);
  const int grp = b * kG + (c >> 4);
  const float v = x[(size_t)b * kC * kN + (size_t)c * kN + n];
  const float hv = (v - mean[grp]) * rstd[grp] * gn_w[c] + gn_b[c];
  h_bnc[idx] = (half_t)hv;   // coalesced write, [b][n][c]
}

// ----------------------- kernel 3: f32 -> f16 convert -----------------------

__global__ __launch_bounds__(256)
void cvt_f16_kernel(const float* __restrict__ src, half_t* __restrict__ dst) {
  const size_t i = (size_t)blockIdx.x * 256 + threadIdx.x;
  dst[i] = (half_t)src[i];
}

// ------------------------- kernel 4: QKV projection -------------------------
// qkv[b][o][n] = sum_c W[o][c] * h[b][n][c] + bias[o]
// 8 waves/block, each one 16x16 tile; block covers 128 o-rows x 16 n-cols.
// q,k stored [b][n][c]; v stored [b][c][n] (AV B-operand layout).

__global__ __launch_bounds__(256)
void qkv_gemm_kernel(const half_t* __restrict__ w16,     // [1536][512]
                     const float*  __restrict__ bias,    // [1536]
                     const half_t* __restrict__ h_bnc,   // [B][N][C]
                     half_t* __restrict__ q_bnc,
                     half_t* __restrict__ k_bnc,
                     half_t* __restrict__ v_bcn) {
  const int wave = threadIdx.x >> 5;
  const int lane = threadIdx.x & 31;
  const int col  = lane & 15;
  const int hi   = lane >> 4;
  const int o0   = blockIdx.x * 128 + wave * 16;
  const int n0   = blockIdx.y * 16;
  const int b    = blockIdx.z;

  const half_t* Arow = w16 + (size_t)(o0 + col) * kC;
  const half_t* Brow = h_bnc + ((size_t)b * kN + n0 + col) * kC;

  v8f acc = {};
  #pragma unroll
  for (int kk = 0; kk < kC / 32; ++kk) {
    v16h a = load_frag_a(Arow, kk * 32, hi);
    v16h bb = load_frag_b(Brow, kk * 32, hi);
    acc = wmma_f16(a, bb, acc);
  }

  #pragma unroll
  for (int r = 0; r < 8; ++r) {
    const int o = o0 + r + 8 * hi;
    const int n = n0 + col;
    const float val = acc[r] + bias[o];
    if (o < kC) {
      q_bnc[((size_t)b * kN + n) * kC + o] = (half_t)val;
    } else if (o < 2 * kC) {
      k_bnc[((size_t)b * kN + n) * kC + (o - kC)] = (half_t)val;
    } else {
      v_bcn[((size_t)b * kC + (o - 2 * kC)) * kN + n] = (half_t)val;
    }
  }
}

// ----------------------- kernel 5: flash attention --------------------------
// One 16-row q tile per block (8 waves). m tiled by 128: wave w computes the
// 16x16 score subtile for columns [mt+16w, mt+16w+16), online softmax is done
// cooperatively in LDS, then each wave accumulates its 64-column slice of AV.

__global__ __launch_bounds__(256)
void attn_flash_kernel(const half_t* __restrict__ q_bnc,
                       const half_t* __restrict__ k_bnc,
                       const half_t* __restrict__ v_bcn,
                       half_t* __restrict__ o_bnc) {
  __shared__ half_t q_tile[16 * kC];     // 16 KB
  __shared__ float  S_lds[16 * 128];     // 8 KB
  __shared__ half_t P_lds[16 * 128];     // 4 KB
  __shared__ float  rowmax[16], rowsum[16], rowfac[16];

  const int tid  = threadIdx.x;
  const int wave = tid >> 5;
  const int lane = tid & 31;
  const int col  = lane & 15;
  const int hi   = lane >> 4;
  const int n0   = blockIdx.x * 16;
  const int b    = blockIdx.y;

  // stage the q tile ([16 rows][512 c], k-contiguous) into LDS
  for (int i = tid; i < 16 * kC / 8; i += 256) {
    const int e = i * 8;
    const int row = e >> 9, c = e & (kC - 1);
    *reinterpret_cast<v8h*>(&q_tile[e]) =
        *reinterpret_cast<const v8h*>(&q_bnc[((size_t)b * kN + n0 + row) * kC + c]);
  }
  if (tid < 16) { rowmax[tid] = -1e30f; rowsum[tid] = 0.f; rowfac[tid] = 0.f; }
  __syncthreads();

  v8f acc0 = {}, acc1 = {}, acc2 = {}, acc3 = {};
  const half_t* Qrow = q_tile + (size_t)col * kC;   // A rows = q rows (M=lane%16)

  for (int mt = 0; mt < kN; mt += 128) {
    // ---- scores: S[n][m] for this wave's 16 m-columns -----------------------
    const int msub = mt + wave * 16;
    const half_t* Krow = k_bnc + ((size_t)b * kN + msub + col) * kC;
    v8f sacc = {};
    #pragma unroll
    for (int kk = 0; kk < kC / 32; ++kk) {
      v16h a = load_frag_a(Qrow, kk * 32, hi);
      v16h bb = load_frag_b(Krow, kk * 32, hi);
      sacc = wmma_f16(a, bb, sacc);
    }
    #pragma unroll
    for (int r = 0; r < 8; ++r)
      S_lds[(r + 8 * hi) * 128 + wave * 16 + col] = sacc[r] * kScale;
    __syncthreads();

    // ---- online softmax over the 128-wide strip (16 rows, one thread each) --
    if (tid < 16) {
      float* srow = &S_lds[tid * 128];
      float newm = rowmax[tid];
      for (int j = 0; j < 128; ++j) newm = fmaxf(newm, srow[j]);
      const float fac = __expf(rowmax[tid] - newm);
      float s = 0.f;
      for (int j = 0; j < 128; ++j) {
        const float p = __expf(srow[j] - newm);
        P_lds[tid * 128 + j] = (half_t)p;
        s += p;
      }
      rowsum[tid] = rowsum[tid] * fac + s;
      rowmax[tid] = newm;
      rowfac[tid] = fac;
    }
    __syncthreads();

    // ---- rescale running AV accumulators -----------------------------------
    float fr[8];
    #pragma unroll
    for (int r = 0; r < 8; ++r) fr[r] = rowfac[r + 8 * hi];
    #pragma unroll
    for (int r = 0; r < 8; ++r) {
      acc0[r] *= fr[r]; acc1[r] *= fr[r]; acc2[r] *= fr[r]; acc3[r] *= fr[r];
    }

    // ---- AV: o[n][c] += P[n][m] * v[m][c], wave owns 64 c columns -----------
    const half_t* Prow = P_lds + (size_t)col * 128;
    const size_t vbase = (size_t)b * kC * kN + (size_t)mt;
    #pragma unroll
    for (int ks = 0; ks < 4; ++ks) {               // K(m) in chunks of 32
      v16h a = load_frag_a(Prow, ks * 32, hi);
      const int c0 = wave * 64 + col;
      v16h b0 = load_frag_b(v_bcn + vbase + (size_t)(c0 +  0) * kN, ks * 32, hi);
      v16h b1 = load_frag_b(v_bcn + vbase + (size_t)(c0 + 16) * kN, ks * 32, hi);
      v16h b2 = load_frag_b(v_bcn + vbase + (size_t)(c0 + 32) * kN, ks * 32, hi);
      v16h b3 = load_frag_b(v_bcn + vbase + (size_t)(c0 + 48) * kN, ks * 32, hi);
      acc0 = wmma_f16(a, b0, acc0);
      acc1 = wmma_f16(a, b1, acc1);
      acc2 = wmma_f16(a, b2, acc2);
      acc3 = wmma_f16(a, b3, acc3);
    }
    __syncthreads();   // protect S_lds/P_lds before next strip
  }

  // ---- finalize: divide by row sums, store o as [b][n][c] f16 ---------------
  float inv[8];
  #pragma unroll
  for (int r = 0; r < 8; ++r) inv[r] = 1.0f / rowsum[r + 8 * hi];
  #pragma unroll
  for (int r = 0; r < 8; ++r) {
    const int n = n0 + r + 8 * hi;
    const size_t base = ((size_t)b * kN + n) * kC + wave * 64 + col;
    o_bnc[base +  0] = (half_t)(acc0[r] * inv[r]);
    o_bnc[base + 16] = (half_t)(acc1[r] * inv[r]);
    o_bnc[base + 32] = (half_t)(acc2[r] * inv[r]);
    o_bnc[base + 48] = (half_t)(acc3[r] * inv[r]);
  }
}

// ---------------- kernel 6: out projection + bias + residual ----------------
// y[b][co][n] = x[b][co][n] + bias[co] + sum_c Wo[co][c] * o[b][n][c]

__global__ __launch_bounds__(256)
void outproj_kernel(const half_t* __restrict__ wo16,    // [512][512]
                    const float*  __restrict__ bias,    // [512]
                    const half_t* __restrict__ o_bnc,   // [B][N][C]
                    const float*  __restrict__ x,       // [B][C][N]
                    float* __restrict__ y) {            // [B][C][N]
  const int wave = threadIdx.x >> 5;
  const int lane = threadIdx.x & 31;
  const int col  = lane & 15;
  const int hi   = lane >> 4;
  const int co0  = blockIdx.x * 128 + wave * 16;
  const int n0   = blockIdx.y * 16;
  const int b    = blockIdx.z;

  const half_t* Arow = wo16 + (size_t)(co0 + col) * kC;
  const half_t* Brow = o_bnc + ((size_t)b * kN + n0 + col) * kC;

  v8f acc = {};
  #pragma unroll
  for (int kk = 0; kk < kC / 32; ++kk) {
    v16h a = load_frag_a(Arow, kk * 32, hi);
    v16h bb = load_frag_b(Brow, kk * 32, hi);
    acc = wmma_f16(a, bb, acc);
  }

  #pragma unroll
  for (int r = 0; r < 8; ++r) {
    const int co = co0 + r + 8 * hi;
    const int n  = n0 + col;
    const size_t idx = ((size_t)b * kC + co) * kN + n;
    y[idx] = x[idx] + bias[co] + acc[r];
  }
}

// ------------------------------- launcher -----------------------------------

extern "C" void kernel_launch(void* const* d_in, const int* in_sizes, int n_in,
                              void* d_out, int out_size, void* d_ws, size_t ws_size,
                              hipStream_t stream) {
  const float* x     = (const float*)d_in[0];
  const float* gn_w  = (const float*)d_in[1];
  const float* gn_b  = (const float*)d_in[2];
  const float* qkv_w = (const float*)d_in[3];
  const float* qkv_b = (const float*)d_in[4];
  const float* out_w = (const float*)d_in[5];
  const float* out_b = (const float*)d_in[6];
  float* y = (float*)d_out;

  // workspace carve-up (all offsets 256B aligned)
  char* ws = (char*)d_ws;
  size_t off = 0;
  auto take = [&](size_t bytes) { void* p = ws + off; off += (bytes + 255) & ~(size_t)255; return p; };
  float*  mean  = (float*)take(kB * kG * sizeof(float));
  float*  rstd  = (float*)take(kB * kG * sizeof(float));
  half_t* wq16  = (half_t*)take((size_t)kO3 * kC * sizeof(half_t));
  half_t* wo16  = (half_t*)take((size_t)kC * kC * sizeof(half_t));
  half_t* h_bnc = (half_t*)take((size_t)kB * kN * kC * sizeof(half_t));
  half_t* q_bnc = (half_t*)take((size_t)kB * kN * kC * sizeof(half_t));
  half_t* k_bnc = (half_t*)take((size_t)kB * kN * kC * sizeof(half_t));
  half_t* v_bcn = (half_t*)take((size_t)kB * kC * kN * sizeof(half_t));
  half_t* o_bnc = (half_t*)take((size_t)kB * kN * kC * sizeof(half_t));

  // 1) group-norm statistics
  gn_stats_kernel<<<kB * kG, 256, 0, stream>>>(x, mean, rstd);
  // 2) normalize + transpose to [b][n][c] f16
  gn_apply_kernel<<<(kB * kC * kN) / 256, 256, 0, stream>>>(x, gn_w, gn_b, mean, rstd, h_bnc);
  // 3) weights to f16
  cvt_f16_kernel<<<(kO3 * kC) / 256, 256, 0, stream>>>(qkv_w, wq16);
  cvt_f16_kernel<<<(kC * kC) / 256, 256, 0, stream>>>(out_w, wo16);
  // 4) QKV projection (WMMA)
  qkv_gemm_kernel<<<dim3(kO3 / 128, kN / 16, kB), 256, 0, stream>>>(
      wq16, qkv_b, h_bnc, q_bnc, k_bnc, v_bcn);
  // 5) flash attention (WMMA QK^T + WMMA AV, online softmax)
  attn_flash_kernel<<<dim3(kN / 16, kB), 256, 0, stream>>>(q_bnc, k_bnc, v_bcn, o_bnc);
  // 6) out projection + bias + residual (WMMA)
  outproj_kernel<<<dim3(kC / 128, kN / 16, kB), 256, 0, stream>>>(
      wo16, out_b, o_bnc, x, y);
}
